// MultiHeadedCrossAttention_7310034338301
// MI455X (gfx1250) — matrix operand order repaired and checked by reading, not compiled, not run on hardware
//
#include <hip/hip_runtime.h>
#include <hip/hip_bf16.h>
#include <stdint.h>

typedef __attribute__((ext_vector_type(16))) _Float16 v16h;
typedef __attribute__((ext_vector_type(8)))  float    v8f;

union Frag16 {
    v16h v;
    _Float16 h[16];
    uint32_t u[8];
};

union HalfVec8 {
    uint4 q;
    _Float16 h[8];
};

__device__ __forceinline__ v8f wmma_f16(const v16h& a, const v16h& b, const v8f& c) {
    return __builtin_amdgcn_wmma_f32_16x16x32_f16(
        /*neg_a=*/false, a, /*neg_b=*/false, b,
        /*c_mod=*/(short)0, c, /*reuse_a=*/false, /*reuse_b=*/false);
}

// -----------------------------------------------------------------------------
// GEMM: C[M,N] = A[M,K](f32) * W[K,N](f32) + bias[N]
// MODE 0: f16 row-major out              (Q projection)
// MODE 1: cols<512 -> f16 row-major K, cols>=512 -> f16 TRANSPOSED vT[d][row]
//         (KV projection; vT gives contiguous PV B-fragments later)
// MODE 2: f32 row-major out              (output projection)
// Block = 256 threads (8 waves), 64x64 tile, K-step 32, double-buffered LDS.
// -----------------------------------------------------------------------------
template <int MODE>
__global__ void __launch_bounds__(256)
mhca_gemm_bias_kernel(const float* __restrict__ A, const float* __restrict__ W,
                      const float* __restrict__ bias,
                      _Float16* __restrict__ outH, _Float16* __restrict__ outT,
                      float* __restrict__ outF,
                      int M, int N, int K) {
    __shared__ _Float16 Ah[2][64][34];   // A tile, row-major [m][k]
    __shared__ _Float16 Wt[2][64][34];   // W tile, transposed [n][k]

    const int tid   = threadIdx.x;
    const int wave  = tid >> 5;
    const int lane  = tid & 31;
    const int grp   = lane >> 4;       // half-wave
    const int l16   = lane & 15;
    const int tm    = wave & 3;        // row sub-tile 0..3
    const int tn    = wave >> 2;       // col sub-tile 0..1 (32 cols each)

    const int nbase = blockIdx.x * 64;
    const int mbase = blockIdx.y * 64;

    v8f c0 = {}, c1 = {};

    const int a_r  = tid >> 2;         // 64 rows, 4 threads/row
    const int a_c0 = (tid & 3) * 8;    // 8 consecutive k
    const int w_k  = tid >> 3;         // 32 k-rows, 8 threads/row
    const int w_n0 = (tid & 7) * 8;    // 8 consecutive n

    const int nsteps = K >> 5;

    // stage tile 0
    {
        const float* arow = A + (size_t)(mbase + a_r) * K + a_c0;
        #pragma unroll
        for (int j = 0; j < 8; ++j) Ah[0][a_r][a_c0 + j] = (_Float16)arow[j];
        const float* wrow = W + (size_t)w_k * N + nbase + w_n0;
        #pragma unroll
        for (int j = 0; j < 8; ++j) Wt[0][w_n0 + j][w_k] = (_Float16)wrow[j];
    }

    for (int step = 0; step < nsteps; ++step) {
        const int cur = step & 1;
        __syncthreads();

        // stage next tile into the other buffer while computing this one
        if (step + 1 < nsteps) {
            const int kb = (step + 1) << 5;
            const float* arow = A + (size_t)(mbase + a_r) * K + kb + a_c0;
            #pragma unroll
            for (int j = 0; j < 8; ++j) Ah[cur ^ 1][a_r][a_c0 + j] = (_Float16)arow[j];
            const float* wrow = W + (size_t)(kb + w_k) * N + nbase + w_n0;
            #pragma unroll
            for (int j = 0; j < 8; ++j) Wt[cur ^ 1][w_n0 + j][w_k] = (_Float16)wrow[j];
        }

        Frag16 a, b0, b1;
        const int Mrow = tm * 16 + l16;
        #pragma unroll
        for (int i = 0; i < 8; ++i) {
            // 16-bit A 16x32 layout: grp0 K in {0..7,16..23}, grp1 +8
            int k0 = ((i < 4) ? 2 * i : 16 + 2 * (i - 4)) + grp * 8;
            a.u[i] = *(const uint32_t*)&Ah[cur][Mrow][k0];
        }
        const int Ncol = tn * 32 + l16;
        #pragma unroll
        for (int i = 0; i < 8; ++i) {
            // 16-bit B 32x16 layout: grp0 K in {0..7,16..23}, grp1 +8
            int kk = ((i < 4) ? 2 * i : 16 + 2 * (i - 4)) + grp * 8;
            b0.u[i] = *(const uint32_t*)&Wt[cur][Ncol][kk];
            b1.u[i] = *(const uint32_t*)&Wt[cur][Ncol + 16][kk];
        }
        c0 = wmma_f16(a.v, b0.v, c0);
        c1 = wmma_f16(a.v, b1.v, c1);
    }

    // Epilogue: C/D layout -> row = r + grp*8 (per VGPR), col = lane&15
    #pragma unroll
    for (int half = 0; half < 2; ++half) {
        v8f c = half ? c1 : c0;
        const int col = nbase + tn * 32 + half * 16 + l16;
        const float bv = bias[col];
        const int row0 = mbase + tm * 16 + grp * 8;

        if constexpr (MODE == 0) {
            #pragma unroll
            for (int r = 0; r < 8; ++r)
                outH[(size_t)(row0 + r) * N + col] = (_Float16)(c[r] + bv);
        } else if constexpr (MODE == 2) {
            #pragma unroll
            for (int r = 0; r < 8; ++r)
                outF[(size_t)(row0 + r) * N + col] = c[r] + bv;
        } else { // MODE == 1: KV split
            if (col < 512) {
                // K half: row-major [row][512]
                #pragma unroll
                for (int r = 0; r < 8; ++r)
                    outH[(size_t)(row0 + r) * 512 + col] = (_Float16)(c[r] + bv);
            } else {
                // V half: transposed vT[d][key], 8 contiguous keys -> one b128
                HalfVec8 pk;
                #pragma unroll
                for (int r = 0; r < 8; ++r) pk.h[r] = (_Float16)(c[r] + bv);
                *(uint4*)&outT[(size_t)(col - 512) * 16384 + row0] = pk.q;
            }
        }
    }
}

// -----------------------------------------------------------------------------
// Attention: one wave per (b, h, 16-row q tile).  B=4, H=8, T1=1024, T2=4096,
// D_K=64.
// Pass A: scores = QK^T/8 (2 WMMAs per 16x16 tile), masked, raw scores stored
//         to the weights output; PER-LANE online max/sum (no cross-lane traffic
//         in the hot loop), merged once with a 4-step butterfly at the end.
// Pass B: normalize scores in place, context = P·V (4 WMMAs per 32-key step),
//         V read from transposed vT[d][key] (contiguous dword fragments).
// -----------------------------------------------------------------------------
__global__ void __launch_bounds__(256)
mhca_attention_kernel(const _Float16* __restrict__ qh,   // [4096][512]
                      const _Float16* __restrict__ kh,   // [16384][512]
                      const _Float16* __restrict__ vT,   // [512][16384]
                      const unsigned char* __restrict__ mask, // [4][4096]
                      float* __restrict__ weights,       // [4][8][1024][4096]
                      float* __restrict__ ctx) {         // [4096][512] f32
    __shared__ float msM[8][16];
    __shared__ float msS[8][16];

    const int tid  = threadIdx.x;
    const int warp = tid >> 5;
    const int lane = tid & 31;
    const int grp  = lane >> 4;
    const int l16  = lane & 15;

    const int wid   = blockIdx.x * 8 + warp;   // 0..2047
    const int b     = wid >> 9;                // / (8*64)
    const int rem   = wid & 511;
    const int h     = rem >> 6;
    const int qbase = (rem & 63) * 16;

    // ---- Q fragments (two K-steps covering D_K=64) ----
    Frag16 qf[2];
    {
        const _Float16* qrow = qh + (size_t)(b * 1024 + qbase + l16) * 512 + h * 64;
        #pragma unroll
        for (int s = 0; s < 2; ++s)
            #pragma unroll
            for (int i = 0; i < 8; ++i) {
                int k0 = s * 32 + ((i < 4) ? 2 * i : 16 + 2 * (i - 4)) + grp * 8;
                qf[s].u[i] = *(const uint32_t*)&qrow[k0];
            }
    }

    // Per-lane online softmax state: slot r covers row (r + grp*8),
    // keys congruent to (lane&15) mod 16.
    float m[8], s[8];
    #pragma unroll
    for (int r = 0; r < 8; ++r) { m[r] = -3.0e38f; s[r] = 0.0f; }

    float* wbase = weights + (size_t)((b * 8 + h) * 1024 + qbase) * 4096;

    // ---- Pass A: scores + per-lane online stats (no cross-lane ops) ----
    for (int t = 0; t < 256; ++t) {
        const int key = t * 16 + l16;
        const _Float16* krow = kh + (size_t)(b * 4096 + key) * 512 + h * 64;
        __builtin_prefetch(krow + 16 * 512, 0, 3);   // next key tile, keep near

        Frag16 bf0, bf1;
        #pragma unroll
        for (int i = 0; i < 8; ++i) {
            int kk = ((i < 4) ? 2 * i : 16 + 2 * (i - 4)) + grp * 8;
            bf0.u[i] = *(const uint32_t*)&krow[kk];
            bf1.u[i] = *(const uint32_t*)&krow[32 + kk];
        }
        v8f c = {};
        c = wmma_f16(qf[0].v, bf0.v, c);
        c = wmma_f16(qf[1].v, bf1.v, c);

        const bool ok = mask[b * 4096 + key] != 0;
        #pragma unroll
        for (int r = 0; r < 8; ++r) {
            float v = ok ? c[r] * 0.125f : -3.0e38f;
            wbase[(size_t)(r + grp * 8) * 4096 + key] = v;   // raw score
            float mn = fmaxf(m[r], v);
            s[r] = s[r] * __expf(m[r] - mn) + __expf(v - mn);
            m[r] = mn;
        }
    }

    // ---- Butterfly merge of (m, s) pairs across the 16-lane half ----
    #pragma unroll
    for (int r = 0; r < 8; ++r) {
        float mm = m[r], ss = s[r];
        #pragma unroll
        for (int d = 1; d < 16; d <<= 1) {
            float mo = __shfl_xor(mm, d, 32);
            float so = __shfl_xor(ss, d, 32);
            float mn = fmaxf(mm, mo);
            ss = ss * __expf(mm - mn) + so * __expf(mo - mn);
            mm = mn;
        }
        m[r] = mm; s[r] = ss;
    }

    // publish per-row max / sum for this wave's 16 q rows
    if (lane == 0) {
        #pragma unroll
        for (int r = 0; r < 8; ++r) { msM[warp][r] = m[r]; msS[warp][r] = s[r]; }
    }
    if (lane == 16) {
        #pragma unroll
        for (int r = 0; r < 8; ++r) { msM[warp][r + 8] = m[r]; msS[warp][r + 8] = s[r]; }
    }
    __syncthreads();
    const float mrow = msM[warp][l16];
    const float inv  = 1.0f / msS[warp][l16];

    // ---- Pass B: normalize weights in place, context = P * V ----
    v8f acc[4] = {};
    float* srow = wbase + (size_t)l16 * 4096;   // this lane's q row (M = lane&15)
    const _Float16* vcol0 = vT + (size_t)(h * 64 + l16) * 16384 + b * 4096;

    for (int step = 0; step < 128; ++step) {
        const int kb = step * 32;

        Frag16 aw;
        #pragma unroll
        for (int i = 0; i < 8; ++i) {
            int k0 = ((i < 4) ? 2 * i : 16 + 2 * (i - 4)) + grp * 8;
            float2 sc = *(float2*)&srow[kb + k0];
            float w0 = __expf(sc.x - mrow) * inv;
            float w1 = __expf(sc.y - mrow) * inv;
            *(float2*)&srow[kb + k0] = make_float2(w0, w1);   // normalized out
            aw.h[2 * i]     = (_Float16)w0;
            aw.h[2 * i + 1] = (_Float16)w1;
        }

        #pragma unroll
        for (int nb = 0; nb < 4; ++nb) {
            // B-fragment from vT: pairs of adjacent keys are contiguous dwords
            const _Float16* vrow = vcol0 + (size_t)(nb * 16) * 16384 + kb;
            Frag16 bf;
            #pragma unroll
            for (int i = 0; i < 8; ++i) {
                int kk = ((i < 4) ? 2 * i : 16 + 2 * (i - 4)) + grp * 8;
                bf.u[i] = *(const uint32_t*)&vrow[kk];
            }
            acc[nb] = wmma_f16(aw.v, bf.v, acc[nb]);
        }
    }

    #pragma unroll
    for (int nb = 0; nb < 4; ++nb)
        #pragma unroll
        for (int r = 0; r < 8; ++r) {
            const int row = b * 1024 + qbase + r + grp * 8;
            const int col = h * 64 + nb * 16 + l16;
            ctx[(size_t)row * 512 + col] = acc[nb][r];
        }
}

// -----------------------------------------------------------------------------
extern "C" void kernel_launch(void* const* d_in, const int* in_sizes, int n_in,
                              void* d_out, int out_size, void* d_ws, size_t ws_size,
                              hipStream_t stream) {
    const float*         query  = (const float*)d_in[0];         // [4,1024,512]
    const float*         memory = (const float*)d_in[1];         // [4,4096,512]
    const unsigned char* mmask  = (const unsigned char*)d_in[2]; // [4,1,4096] bool
    const float*         Wq     = (const float*)d_in[3];
    const float*         bq     = (const float*)d_in[4];
    const float*         Wvk    = (const float*)d_in[5];
    const float*         bvk    = (const float*)d_in[6];
    const float*         Wo     = (const float*)d_in[7];
    const float*         bo     = (const float*)d_in[8];

    float* out_ctx = (float*)d_out;                          // 4*1024*512
    float* out_w   = (float*)d_out + (size_t)4 * 1024 * 512; // 4*8*1024*4096

    char* ws = (char*)d_ws;
    _Float16* qh  = (_Float16*)ws;                                   //  4 MiB
    _Float16* kv  = (_Float16*)(ws + (size_t)4096 * 512 * 2);        // 16 MiB (K)
    _Float16* vT  = (_Float16*)(ws + (size_t)4096 * 512 * 2
                                   + (size_t)16384 * 512 * 2);       // 16 MiB (V^T)
    float*    ctx = (float*)(ws + (size_t)4096 * 512 * 2
                                + (size_t)16384 * 512 * 2
                                + (size_t)512 * 16384 * 2);          //  8 MiB

    dim3 blk(256);

    // q = query @ Wq + bq -> f16        (M=4096, N=512, K=512)
    mhca_gemm_bias_kernel<0><<<dim3(512 / 64, 4096 / 64), blk, 0, stream>>>(
        query, Wq, bq, qh, nullptr, nullptr, 4096, 512, 512);

    // kv = memory @ Wvk + bvk -> f16    (M=16384, N=1024, K=512)
    // cols 0..511 -> K row-major, cols 512..1023 -> V transposed
    mhca_gemm_bias_kernel<1><<<dim3(1024 / 64, 16384 / 64), blk, 0, stream>>>(
        memory, Wvk, bvk, kv, vT, nullptr, 16384, 1024, 512);

    // attention: 2048 waves = 256 blocks x 8 waves
    mhca_attention_kernel<<<dim3(256), blk, 0, stream>>>(
        qh, kv, vT, mmask, out_w, ctx);

    // context @ Wo + bo -> f32 output   (M=4096, N=512, K=512)
    mhca_gemm_bias_kernel<2><<<dim3(512 / 64, 4096 / 64), blk, 0, stream>>>(
        ctx, Wo, bo, nullptr, nullptr, out_ctx, 4096, 512, 512);
}
